// QpNet_57870389346828
// MI455X (gfx1250) — matrix operand order repaired and checked by reading, not compiled, not run on hardware
//
#include <hip/hip_runtime.h>
#include <stdint.h>

// ---------------- problem constants ----------------
#define NB        16384
#define NI        32
#define NH        1024
#define NU        10
#define MCON      138
#define MPAD      160
#define IPM_ITERS 30
#define NEG_SLOPE 0.2f
#define SIGMA     0.1f
#define BN_EPS    1e-5f
#define QP_EPS    1e-4f

typedef __bf16 bf16;
typedef bf16  v16bf __attribute__((ext_vector_type(16)));
typedef bf16  v2bf  __attribute__((ext_vector_type(2)));
typedef float v2f   __attribute__((ext_vector_type(2)));
typedef float v8f   __attribute__((ext_vector_type(8)));
typedef unsigned int v4u __attribute__((ext_vector_type(4)));
typedef int   v8i   __attribute__((ext_vector_type(8)));
typedef int   v4i   __attribute__((ext_vector_type(4)));

// ---------------- workspace layout (float indices) ----------------
#define OFF_A2C2   0            // 32
#define OFF_QM     32           // 256   : Q_hat zero-padded to 16x16
#define OFF_GROWS  288          // 2560  : G rows, [160][16] zero-padded
#define OFF_GT     2848         // 2560  : G^T,    [16][160] zero-padded
#define OFF_HVEC   5408         // 160
#define OFF_BIAS2  5568         // 16    : FC2 bias with BN1 shift folded in
#define OFF_PART2  6608         // 1024*32 : per-wave BN2 partial stats
#define OFF_W2P    39376        // 8192 floats = [16][1024] bf16 scaled FC2 weights
#define OFF_PART1  47568        // 1024*2048 : per-wave BN1 partial stats
#define OFF_PPRE   2144720      // 16384*16  : pre-BN2 FC2 activations
#define OFF_H1     2406864      // 16384*1024 bf16 (8388608 floats)

// A-fragment K position for 16-bit 16x32 WMMA (contiguous order, k = 16*half + 2v)
__device__ __forceinline__ int klayA(int v, int hh) {
  return 16 * hh + 2 * v;
}

// ---------------- Tensor Data Mover: 2D f32 tile -> LDS ----------------
// 6-arg builtin flavor (clang-23 / therock-10.0 headers):
//   (uint32x4 group0, int32x8 group1, int32x4, int32x4, int32x8, i32 cpol)
__device__ __forceinline__ void tdm_load_2d_f32(const void* gsrc, unsigned lds_byte,
                                                unsigned tile_w, unsigned tile_h,
                                                unsigned stride_elems,
                                                unsigned tensor_w, unsigned tensor_h) {
  unsigned long long ga = (unsigned long long)(uintptr_t)gsrc;
  v4u g0 = { 1u,                                            // count=1, user D#
             lds_byte,                                      // lds_addr
             (unsigned)(ga & 0xffffffffull),                // global_addr[31:0]
             (unsigned)((ga >> 32) & 0x1ffffffull) | (2u << 30) };  // addr[56:32] | type=2
  unsigned w0 = 2u << 16;                                   // data_size = 4 bytes
  unsigned w1 = (tensor_w & 0xffffu) << 16;                 // tensor_dim0[15:0]
  unsigned w2 = ((tensor_w >> 16) & 0xffffu) | ((tensor_h & 0xffffu) << 16);
  unsigned w3 = ((tensor_h >> 16) & 0xffffu) | ((tile_w & 0xffffu) << 16);
  unsigned w4 = (tile_h & 0xffffu);                         // tile_dim1, tile_dim2=0
  unsigned w5 = stride_elems;                               // tensor_dim0_stride[31:0]
  v8i g1 = { (int)w0, (int)w1, (int)w2, (int)w3, (int)w4, (int)w5, 0, 0 };
  v4i gz4 = { 0, 0, 0, 0 };
  v8i gz8 = { 0, 0, 0, 0, 0, 0, 0, 0 };
  __builtin_amdgcn_tensor_load_to_lds(g0, g1, gz4, gz4, gz8, 0);
}

__device__ __forceinline__ unsigned lds_off(const void* p) {
  return (unsigned)(unsigned long long)(uintptr_t)p;   // low 32 bits of flat LDS addr
}

// =====================================================================
// K1: h1 = leaky(x @ fc1_w^T + b1), bf16 WMMA (K=32 == one 16x16x32).
//     Stores h1 in bf16 (already A-fragment friendly) + per-wave BN1 stats.
// =====================================================================
__global__ __launch_bounds__(256) void k_fc1(const float* __restrict__ x,
                                             const float* __restrict__ w1g,
                                             const float* __restrict__ b1,
                                             float* __restrict__ part1,
                                             bf16* __restrict__ h1) {
  __shared__ float xt[128 * 32];                 // 128-row batch slab of x
  const int t = threadIdx.x, wave = t >> 5, lane = t & 31;
  const int hh = lane >> 4, nn = lane & 15;

  if (wave == 0) {                               // TDM: 128x32 f32 tile -> LDS
    tdm_load_2d_f32(x + (size_t)blockIdx.x * 128 * 32, lds_off(xt),
                    32u, 128u, 32u, 32u, 128u);
    __builtin_amdgcn_s_wait_tensorcnt(0);
  }
  __syncthreads();

  // A fragment (this wave's 16 batch rows), built once: K=32 exactly
  v16bf a;
#pragma unroll
  for (int v = 0; v < 8; ++v) {
    int k = klayA(v, hh);
    a[2 * v]     = (bf16)xt[(wave * 16 + nn) * 32 + k];
    a[2 * v + 1] = (bf16)xt[(wave * 16 + nn) * 32 + k + 1];
  }

  for (int n = 0; n < 64; ++n) {                 // 64 column tiles of NH=1024
    v16bf bfr;
#pragma unroll
    for (int v = 0; v < 8; ++v) {                // B[k][n'] = fc1_w[16n+n'][k]
      int k = 16 * hh + 2 * v;
      v2f wp = *(const v2f*)(w1g + (size_t)(n * 16 + nn) * 32 + k);
      bfr[2 * v] = (bf16)wp.x; bfr[2 * v + 1] = (bf16)wp.y;
    }
    v8f c = {0.f, 0.f, 0.f, 0.f, 0.f, 0.f, 0.f, 0.f};
    c = __builtin_amdgcn_wmma_f32_16x16x32_bf16(false, a, false, bfr,
                                                (short)0, c, false, false);
    const int col = n * 16 + nn;
    const float bias = b1[col];
    float ssum = 0.f, sq = 0.f;
#pragma unroll
    for (int r = 0; r < 8; ++r) {                // D row M = r + 8*half
      float y = c[r] + bias;
      y = (y > 0.f) ? y : NEG_SLOPE * y;
      int row = blockIdx.x * 128 + wave * 16 + r + 8 * hh;
      h1[(size_t)row * NH + col] = (bf16)y;
      ssum += y; sq += y * y;
    }
    ssum += __shfl_xor(ssum, 16, 32);
    sq   += __shfl_xor(sq, 16, 32);
    if (hh == 0) {                               // deterministic per-wave partials
      size_t base = ((size_t)blockIdx.x * 8 + wave) * 2048;
      part1[base + col] = ssum;
      part1[base + 1024 + col] = sq;
    }
  }
}

// =====================================================================
// K2: BN1 stats -> fold scale into FC2 weights (bf16), deterministic
//     tree-reduce the BN1 shift into the FC2 bias.
// =====================================================================
__global__ __launch_bounds__(1024) void k_fold(const float* __restrict__ part1,
                                               const float* __restrict__ fc2_w,
                                               const float* __restrict__ fc2_b,
                                               const float* __restrict__ g1v,
                                               const float* __restrict__ b1v,
                                               bf16* __restrict__ W2p,
                                               float* __restrict__ bias2p) {
  __shared__ float red[1024];
  const int k = threadIdx.x;
  float s = 0.f, q = 0.f;
  for (int wg = 0; wg < 1024; ++wg) {            // fixed-order reduction
    s += part1[(size_t)wg * 2048 + k];
    q += part1[(size_t)wg * 2048 + 1024 + k];
  }
  const float mean = s * (1.f / NB);
  const float var  = q * (1.f / NB) - mean * mean;
  const float aa   = g1v[k] * rsqrtf(var + BN_EPS);
  const float cc   = b1v[k] - mean * aa;
#pragma unroll
  for (int u = 0; u < 16; ++u) {
    float wv = (u < NU) ? fc2_w[u * NH + k] : 0.f;
    W2p[u * NH + k] = (bf16)(wv * aa);
  }
  for (int u = 0; u < NU; ++u) {                 // bias' = b2 + sum_k W2[u,k]*c[k]
    red[k] = fc2_w[u * NH + k] * cc;
    __syncthreads();
    for (int off = 512; off > 0; off >>= 1) {
      if (k < off) red[k] += red[k + off];
      __syncthreads();
    }
    if (k == 0) bias2p[u] = red[0] + fc2_b[u];
    __syncthreads();
  }
  if (k >= NU && k < 16) bias2p[k] = 0.f;
}

// =====================================================================
// K3: p_pre = leaky(h1_bf16 @ W2p^T + bias'), 32 chained bf16 WMMAs,
//     per-wave BN2 partial stats.
// =====================================================================
__global__ __launch_bounds__(256) void k_fc2(const bf16* __restrict__ h1,
                                             const bf16* __restrict__ W2p,
                                             const float* __restrict__ bias2p,
                                             float* __restrict__ p_pre,
                                             float* __restrict__ part2) {
  const int t = threadIdx.x, wave = t >> 5, lane = t & 31;
  const int hh = lane >> 4, nn = lane & 15;
  const int mt = blockIdx.x * 8 + wave;

  v8f c = {0.f, 0.f, 0.f, 0.f, 0.f, 0.f, 0.f, 0.f};
  for (int kt = 0; kt < 32; ++kt) {
    v16bf a, b;
#pragma unroll
    for (int v = 0; v < 8; ++v) {
      int ka = 32 * kt + klayA(v, hh);
      v2bf pa = *(const v2bf*)(h1 + (size_t)(mt * 16 + nn) * NH + ka);
      a[2 * v] = pa.x; a[2 * v + 1] = pa.y;
      int kb = 32 * kt + 16 * hh + 2 * v;
      v2bf pb = *(const v2bf*)(W2p + (size_t)nn * NH + kb);
      b[2 * v] = pb.x; b[2 * v + 1] = pb.y;
    }
    c = __builtin_amdgcn_wmma_f32_16x16x32_bf16(false, a, false, b,
                                                (short)0, c, false, false);
  }
  const float bias = bias2p[nn];
  float ssum = 0.f, sq = 0.f;
#pragma unroll
  for (int r = 0; r < 8; ++r) {
    float y = c[r] + bias;
    y = (y > 0.f) ? y : NEG_SLOPE * y;
    p_pre[(size_t)(mt * 16 + r + 8 * hh) * 16 + nn] = y;
    ssum += y; sq += y * y;
  }
  ssum += __shfl_xor(ssum, 16, 32);
  sq   += __shfl_xor(sq, 16, 32);
  if (hh == 0) {
    size_t base = ((size_t)blockIdx.x * 8 + wave) * 32;
    part2[base + nn] = ssum;
    part2[base + 16 + nn] = sq;
  }
}

// =====================================================================
// K4: build QP (Q_hat, G, h) + BN2 affine params. Single small block.
// =====================================================================
__global__ __launch_bounds__(256) void k_qp(const float* __restrict__ part2,
                                            const float* __restrict__ bn2_g,
                                            const float* __restrict__ bn2_b,
                                            const float* __restrict__ Lq,
                                            const float* __restrict__ Lp,
                                            const float* __restrict__ Lr,
                                            const float* __restrict__ Am,
                                            const float* __restrict__ Bm,
                                            const float* __restrict__ u0,
                                            const float* __restrict__ s0,
                                            float* __restrict__ a2c2,
                                            float* __restrict__ Qm16,
                                            float* __restrict__ Grows,
                                            float* __restrict__ GT,
                                            float* __restrict__ hvec) {
  __shared__ float Qs[1024], Ps[1024], Rs[4], Bh[1280], Tm[1280], pw[4][64];
  const int t = threadIdx.x;
  for (int idx = t; idx < 1024; idx += 256) {    // Q = tril(L)tril(L)^T + eps I
    int ii = idx >> 5, jj = idx & 31;
    int km = ii < jj ? ii : jj;
    float sq = 0.f, sp = 0.f;
    for (int k = 0; k <= km; ++k) {
      sq += Lq[ii * 32 + k] * Lq[jj * 32 + k];
      sp += Lp[ii * 32 + k] * Lp[jj * 32 + k];
    }
    if (ii == jj) { sq += QP_EPS; sp += QP_EPS; }
    Qs[idx] = sq; Ps[idx] = sp;
  }
  if (t < 4) {
    int ii = t >> 1, jj = t & 1, km = ii < jj ? ii : jj;
    float s = 0.f;
    for (int k = 0; k <= km; ++k) s += Lr[ii * 2 + k] * Lr[jj * 2 + k];
    if (ii == jj) s += QP_EPS;
    Rs[t] = s;
  }
  if (t < 64) pw[0][t] = Bm[t];                  // A^0 B
  __syncthreads();
  for (int s = 1; s < 4; ++s) {                  // A^s B
    if (t < 64) {
      int ii = t >> 1, cc = t & 1;
      float acc = 0.f;
      for (int k = 0; k < 32; ++k) acc += Am[ii * 32 + k] * pw[s - 1][k * 2 + cc];
      pw[s][t] = acc;
    }
    __syncthreads();
  }
  for (int idx = t; idx < 1280; idx += 256) {    // B_hat [128][10]
    int r = idx / 10, cc = idx % 10;
    int bi = r >> 5, bj = cc >> 1, ccc = cc & 1;
    Bh[idx] = (bj <= bi) ? pw[bi - bj][(r & 31) * 2 + ccc] : 0.f;
  }
  __syncthreads();
  for (int idx = t; idx < 1280; idx += 256) {    // T = Q_diag @ B_hat
    int r = idx / 10, cc = idx % 10, blk = r >> 5;
    const float* Mb = (blk < 3) ? Qs : Ps;
    float acc = 0.f;
    for (int k = 0; k < 32; ++k) acc += Mb[(r & 31) * 32 + k] * Bh[(blk * 32 + k) * 10 + cc];
    Tm[idx] = acc;
  }
  __syncthreads();
  {                                              // Q_hat (padded 16x16)
    int ii = t >> 4, jj = t & 15;
    float v = 0.f;
    if (ii < NU && jj < NU) {
      for (int k = 0; k < 128; ++k) v += Bh[k * 10 + ii] * Tm[k * 10 + jj];
      if ((ii >> 1) == (jj >> 1)) v += Rs[(ii & 1) * 2 + (jj & 1)];
    }
    Qm16[t] = v;
  }
  for (int idx = t; idx < 2560; idx += 256) {    // G rows + G^T (zero padded)
    int m = idx >> 4, ii = idx & 15;
    float g = 0.f;
    if (ii < NU && m < MCON)
      g = (m < NU) ? ((m == ii) ? 1.f : 0.f) : Bh[(m - NU) * 10 + ii];
    Grows[m * 16 + ii] = g;
    GT[ii * MPAD + m] = g;
  }
  for (int idx = t; idx < MPAD; idx += 256) {    // h = G u0 + s0
    float hv = 0.f;
    if (idx < MCON) {
      float acc = 0.f;
      if (idx < NU) acc = u0[idx];
      else for (int k = 0; k < NU; ++k) acc += Bh[(idx - NU) * 10 + k] * u0[k];
      hv = acc + s0[idx];
    }
    hvec[idx] = hv;
  }
  if (t < 16) {                                  // BN2 affine (deterministic)
    float a = 0.f, c = 0.f;
    if (t < NU) {
      float s = 0.f, q = 0.f;
      for (int wg = 0; wg < 1024; ++wg) {
        s += part2[wg * 32 + t];
        q += part2[wg * 32 + 16 + t];
      }
      float mean = s * (1.f / NB);
      float var = q * (1.f / NB) - mean * mean;
      a = bn2_g[t] * rsqrtf(var + BN_EPS);
      c = bn2_b[t] - mean * a;
    }
    a2c2[t] = a; a2c2[16 + t] = c;
  }
}

// =====================================================================
// K5: fused 30-iteration primal-dual IPM. One wave per batch element.
//     H = Q_hat + G^T diag(lam/s) G via 5 chained bf16 WMMAs (K=160),
//     10x10 solve = wave-cooperative Gauss-Jordan with shfl pivots.
// =====================================================================
__global__ __launch_bounds__(256) void k_ipm(const float* __restrict__ Qm16,
                                             const float* __restrict__ Grows,
                                             const float* __restrict__ GTg,
                                             const float* __restrict__ hvec,
                                             const float* __restrict__ p_pre,
                                             const float* __restrict__ a2c2,
                                             float* __restrict__ out) {
  __shared__ float GTl[16 * MPAD];               // G^T, f32, for A-frag staging
  __shared__ float wb[8 * MPAD];                 // per-wave lam/s vector
  __shared__ float Hb[8 * 256];                  // per-wave 16x16 H dump
  const int t = threadIdx.x, wave = t >> 5, lane = t & 31;
  const int hh = lane >> 4, nn = lane & 15;

  for (int j = t; j < 8 * MPAD; j += 256) wb[j] = 0.f;   // padded K slots stay 0
  if (wave == 0) {
    tdm_load_2d_f32(GTg, lds_off(GTl), 16u * MPAD, 1u, 16u * MPAD, 16u * MPAD, 1u);
    __builtin_amdgcn_s_wait_tensorcnt(0);
  }
  __syncthreads();

  const int b = blockIdx.x * 8 + wave;

  float gr[5][10], hr[5], sv[5], lv[5];
  bool vld[5];
#pragma unroll
  for (int r = 0; r < 5; ++r) {
    int m = lane + 32 * r;                       // m < 160, arrays padded
    vld[r] = (m < MCON);
#pragma unroll
    for (int c = 0; c < 10; ++c) gr[r][c] = Grows[m * 16 + c];
    hr[r] = hvec[m];
    sv[r] = 1.f; lv[r] = 1.f;                    // s_init = lam_init = 1
  }
  float q[10];
#pragma unroll
  for (int c = 0; c < 10; ++c) q[c] = (lane < NU) ? Qm16[(lane & 15) * 16 + c] : 0.f;
  float ppr = p_pre[(size_t)b * 16 + (lane & 15)];
  float pj = (lane < NU) ? ppr * a2c2[lane & 15] + a2c2[16 + (lane & 15)] : 0.f;

  v8f cq;                                        // Q_hat in C/D layout (WMMA seed)
#pragma unroll
  for (int r = 0; r < 8; ++r) cq[r] = Qm16[(r + 8 * hh) * 16 + nn];

  v16bf Bf[5];                                   // B = G (160x16), loop-invariant
#pragma unroll
  for (int tt = 0; tt < 5; ++tt)
#pragma unroll
    for (int v = 0; v < 8; ++v) {
      int k = 32 * tt + 16 * hh + 2 * v;
      Bf[tt][2 * v]     = (bf16)GTl[nn * MPAD + k];
      Bf[tt][2 * v + 1] = (bf16)GTl[nn * MPAD + k + 1];
    }

  float z[10];
#pragma unroll
  for (int c = 0; c < 10; ++c) z[c] = 0.f;
  float* wbm = wb + wave * MPAD;
  float* Hbm = Hb + wave * 256;

  for (int it = 0; it < IPM_ITERS; ++it) {
    float rp[5], rc[5], tv[5];
    float sl = 0.f;
#pragma unroll
    for (int r = 0; r < 5; ++r) {                // rp = Gz + s - h  (rows)
      float d = sv[r] - hr[r];
#pragma unroll
      for (int c = 0; c < 10; ++c) d += gr[r][c] * z[c];
      rp[r] = d;
      if (vld[r]) sl += sv[r] * lv[r];
    }
#pragma unroll
    for (int off = 1; off < 32; off <<= 1) sl += __shfl_xor(sl, off, 32);
    const float musig = SIGMA * sl * (1.f / MCON);
#pragma unroll
    for (int r = 0; r < 5; ++r) {
      float rcv = sv[r] * lv[r] - musig;
      float is = 1.f / sv[r];
      rc[r] = rcv;
      float wr = vld[r] ? lv[r] * is : 0.f;
      tv[r] = vld[r] ? (lv[r] * rp[r] - rcv) * is : 0.f;
      if (vld[r]) wbm[lane + 32 * r] = wr;       // w = lam/s for WMMA staging
    }
    float colp[10];                              // (lam + t)^T G, replicated
#pragma unroll
    for (int c = 0; c < 10; ++c) {
      float acc = 0.f;
#pragma unroll
      for (int r = 0; r < 5; ++r) acc += (lv[r] + tv[r]) * gr[r][c];
#pragma unroll
      for (int off = 1; off < 32; off <<= 1) acc += __shfl_xor(acc, off, 32);
      colp[c] = acc;
    }
    __syncthreads();                             // wb visible to whole wave

    v8f accH = cq;                               // H = Q_hat + (wG)^T G
#pragma unroll
    for (int tt = 0; tt < 5; ++tt) {
      v16bf af;
#pragma unroll
      for (int v = 0; v < 8; ++v) {
        int k = 32 * tt + klayA(v, hh);
        af[2 * v]     = (bf16)(wbm[k]     * GTl[nn * MPAD + k]);
        af[2 * v + 1] = (bf16)(wbm[k + 1] * GTl[nn * MPAD + k + 1]);
      }
      accH = __builtin_amdgcn_wmma_f32_16x16x32_bf16(false, af, false, Bf[tt],
                                                     (short)0, accH, false, false);
    }
#pragma unroll
    for (int r = 0; r < 8; ++r) Hbm[(r + 8 * hh) * 16 + nn] = accH[r];
    __syncthreads();

    // ---- wave-cooperative Gauss-Jordan: lane j owns row j of H ----
    float h[10], rhs;
    {
      float qz = 0.f;
#pragma unroll
      for (int c = 0; c < 10; ++c) {
        h[c] = Hbm[(lane & 15) * 16 + c];
        qz += q[c] * z[c];
      }
      float colj = 0.f;
#pragma unroll
      for (int c = 0; c < 10; ++c) colj = (lane == c) ? colp[c] : colj;
      rhs = -(qz + pj + colj);                   // rhs = -(rd + t^T G)
    }
#pragma unroll
    for (int k = 0; k < 10; ++k) {
      float pr = __shfl(rhs, k, 32);
      float pk[10];
#pragma unroll
      for (int c = 0; c < 10; ++c) pk[c] = (c >= k) ? __shfl(h[c], k, 32) : 0.f;
      float inv = 1.f / pk[k];
      float f = (lane == k) ? 0.f : h[k] * inv;
#pragma unroll
      for (int c = 0; c < 10; ++c) if (c >= k) h[c] -= f * pk[c];
      rhs -= f * pr;
    }
    float diag = 1.f;
#pragma unroll
    for (int c = 0; c < 10; ++c) diag = (lane == c) ? h[c] : diag;
    float dzl = rhs / diag;
    float dz[10];
#pragma unroll
    for (int c = 0; c < 10; ++c) dz[c] = __shfl(dzl, c, 32);

    // ---- ds, dlam, fraction-to-boundary alpha ----
    float amin = 1.0e30f;
    float dsv[5], dlv[5];
#pragma unroll
    for (int r = 0; r < 5; ++r) {
      float d = -rp[r];
#pragma unroll
      for (int c = 0; c < 10; ++c) d -= gr[r][c] * dz[c];
      dsv[r] = d;
      float dl = (-rc[r] - lv[r] * d) / sv[r];
      dlv[r] = dl;
      if (vld[r]) {
        if (d  < 0.f) amin = fminf(amin, -sv[r] / d);
        if (dl < 0.f) amin = fminf(amin, -lv[r] / dl);
      }
    }
#pragma unroll
    for (int off = 1; off < 32; off <<= 1) amin = fminf(amin, __shfl_xor(amin, off, 32));
    const float alpha = fminf(1.f, 0.99f * amin);
#pragma unroll
    for (int c = 0; c < 10; ++c) z[c] += alpha * dz[c];
#pragma unroll
    for (int r = 0; r < 5; ++r) if (vld[r]) {
      sv[r] += alpha * dsv[r];
      lv[r] += alpha * dlv[r];
    }
  }
  if (lane == 0) out[b] = z[0];                  // output = z[:,0]
}

// =====================================================================
extern "C" void kernel_launch(void* const* d_in, const int* in_sizes, int n_in,
                              void* d_out, int out_size, void* d_ws, size_t ws_size,
                              hipStream_t stream) {
  (void)in_sizes; (void)n_in; (void)out_size; (void)ws_size;
  const float* x    = (const float*)d_in[0];
  const float* w1   = (const float*)d_in[1];
  const float* b1   = (const float*)d_in[2];
  const float* w2   = (const float*)d_in[3];
  const float* b2   = (const float*)d_in[4];
  const float* bn1g = (const float*)d_in[5];
  const float* bn1b = (const float*)d_in[6];
  const float* bn2g = (const float*)d_in[7];
  const float* bn2b = (const float*)d_in[8];
  const float* Lq   = (const float*)d_in[9];
  const float* Lp   = (const float*)d_in[10];
  const float* Lr   = (const float*)d_in[11];
  const float* Am   = (const float*)d_in[12];
  const float* Bm   = (const float*)d_in[13];
  const float* u0   = (const float*)d_in[14];
  const float* s0   = (const float*)d_in[15];

  float* f      = (float*)d_ws;
  float* a2c2   = f + OFF_A2C2;
  float* Qm16   = f + OFF_QM;
  float* Grows  = f + OFF_GROWS;
  float* GT     = f + OFF_GT;
  float* hvec   = f + OFF_HVEC;
  float* bias2p = f + OFF_BIAS2;
  float* part2  = f + OFF_PART2;
  bf16*  W2p    = (bf16*)(f + OFF_W2P);
  float* part1  = f + OFF_PART1;
  float* p_pre  = f + OFF_PPRE;
  bf16*  h1     = (bf16*)(f + OFF_H1);

  k_fc1 <<<128,  256, 0, stream>>>(x, w1, b1, part1, h1);
  k_fold<<<1,   1024, 0, stream>>>(part1, w2, b2, bn1g, bn1b, W2p, bias2p);
  k_fc2 <<<128,  256, 0, stream>>>(h1, W2p, bias2p, p_pre, part2);
  k_qp  <<<1,    256, 0, stream>>>(part2, bn2g, bn2b, Lq, Lp, Lr, Am, Bm, u0, s0,
                                   a2c2, Qm16, Grows, GT, hvec);
  k_ipm <<<2048, 256, 0, stream>>>(Qm16, Grows, GT, hvec, p_pre, a2c2, (float*)d_out);
}
